// MambaSpike_25855703121988
// MI455X (gfx1250) — matrix operand order repaired and checked by reading, not compiled, not run on hardware
//
#include <hip/hip_runtime.h>
#include <hip/hip_bf16.h>

// ---------------- types for WMMA ----------------
typedef _Float16 v16h __attribute__((ext_vector_type(16)));
typedef _Float16 v8h  __attribute__((ext_vector_type(8)));
typedef float    v8f  __attribute__((ext_vector_type(8)));

#define Bn   64
#define Tn   128
#define HCn  32
#define SCn  64
#define DM   128
#define DI   256
#define DS   16
#define DTR  8
#define MTOT (Bn * Tn)        // 8192 rows
#define SFEAT 4096            // 64*8*8

// ---------------- gfx1250 async-to-LDS path (guarded) ----------------
#if defined(__has_builtin)
#if __has_builtin(__builtin_amdgcn_global_load_async_to_lds_b128) && \
    __has_builtin(__builtin_amdgcn_s_wait_asynccnt)
#define HAVE_ASYNC_LDS 1
#endif
#endif
#ifndef HAVE_ASYNC_LDS
#define HAVE_ASYNC_LDS 0
#endif

#if HAVE_ASYNC_LDS
// Builtin signature (from hipcc diagnostic): param0 = int4 addrspace(1)*,
// param1 = int4 addrspace(3)*, then imm offset, imm cpol.
typedef int v4i_ __attribute__((vector_size(16)));
typedef __attribute__((address_space(1))) v4i_ GAS1v4i;
typedef __attribute__((address_space(3))) v4i_ LAS3v4i;
#endif

// Stage nVec4 16-byte vectors from global to LDS (async when available).
__device__ __forceinline__ void stage_to_lds_16B(float* ldsDst, const float* gSrc,
                                                 int nVec4, int tid, int nthr) {
#if HAVE_ASYNC_LDS
    for (int i = tid; i < nVec4; i += nthr) {
        __builtin_amdgcn_global_load_async_to_lds_b128(
            (GAS1v4i*)(void*)(gSrc + 4 * i),
            (LAS3v4i*)(void*)(ldsDst + 4 * i), 0, 0);
    }
    __builtin_amdgcn_s_wait_asynccnt(0);
#else
    for (int i = tid; i < nVec4; i += nthr) {
        float4 v = *(const float4*)(gSrc + 4 * i);
        *(float4*)(ldsDst + 4 * i) = v;
    }
#endif
}

// ============================================================
// f32 -> f16 converter (for weight staging)
// ============================================================
__global__ void cvt_f32_f16(const float* __restrict__ src, _Float16* __restrict__ dst, int n) {
    int i = blockIdx.x * blockDim.x + threadIdx.x;
    if (i < n) dst[i] = (_Float16)src[i];
}

// ============================================================
// Spiking frontend: 1 workgroup (512 thr) per batch element, T-loop inside.
// x_t tile staged into LDS (async) each step; s1 double-buffered in scratch;
// s2/s3/tmp in LDS. LDS total: (2312+2048+4096+4096)*4 = 50208 B (49.0 KiB).
// ============================================================
__global__ void __launch_bounds__(512)
frontend_kernel(
    const float* __restrict__ x,
    const float* __restrict__ c1w, const float* __restrict__ c1b,
    const float* __restrict__ c2w, const float* __restrict__ c2b,
    const float* __restrict__ c3w, const float* __restrict__ c3b,
    const float* __restrict__ r1w, const float* __restrict__ r1b,
    const float* __restrict__ r2w, const float* __restrict__ r2b,
    const float* __restrict__ r3w, const float* __restrict__ r3b,
    float* __restrict__ s1ga, float* __restrict__ s1gb, float* __restrict__ mem1g,
    float* __restrict__ mem2g, float* __restrict__ mem3g,
    float* __restrict__ spikes)
{
    __shared__ float xts[2312];   // 2*34*34 input tile
    __shared__ float s2[2048];    // 32*8*8
    __shared__ float s3[4096];    // 64*8*8
    __shared__ float tmp[4096];   // stage2/3 currents

    const int b   = blockIdx.x;
    const int tid = threadIdx.x;
    const int NT_ = 512;

    float* s1a = s1ga  + (size_t)b * 9248;
    float* s1b = s1gb  + (size_t)b * 9248;
    float* m1  = mem1g + (size_t)b * 9248;
    float* m2  = mem2g + (size_t)b * 2048;
    float* m3  = mem3g + (size_t)b * 4096;

    for (int i = tid; i < 9248; i += NT_) { s1a[i] = 0.f; s1b[i] = 0.f; m1[i] = 0.f; }
    for (int i = tid; i < 2048; i += NT_) { s2[i] = 0.f; m2[i] = 0.f; }
    for (int i = tid; i < 4096; i += NT_) { s3[i] = 0.f; m3[i] = 0.f; }
    __syncthreads();

    for (int t = 0; t < Tn; ++t) {
        const float* xt = x + (size_t)(b * Tn + t) * 2312;  // 2*34*34
        float* s1p = (t & 1) ? s1b : s1a;   // previous spikes
        float* s1c = (t & 1) ? s1a : s1b;   // current spikes

        // stage x_t tile into LDS (async-to-LDS when available)
        stage_to_lds_16B(xts, xt, 2312 / 4, tid, NT_);
        __syncthreads();

        // ---- stage 1 (fused conv1 -> pool -> recur -> LIF): writes s1c ----
        for (int idx = tid; idx < 9248; idx += NT_) {
            int c = idx / 289, rem = idx % 289, i = rem / 17, j = rem % 17;
            float mx = -1e30f;
            for (int di = 0; di < 2; ++di)
            for (int dj = 0; dj < 2; ++dj) {
                int oy = 2 * i + di, ox = 2 * j + dj;
                float acc = c1b[c];
                for (int ci = 0; ci < 2; ++ci)
                for (int ky = 0; ky < 3; ++ky) {
                    int yy = oy + ky - 1; if ((unsigned)yy >= 34u) continue;
                    for (int kx = 0; kx < 3; ++kx) {
                        int xx = ox + kx - 1; if ((unsigned)xx >= 34u) continue;
                        acc += xts[ci * 1156 + yy * 34 + xx] * c1w[((c * 2 + ci) * 3 + ky) * 3 + kx];
                    }
                }
                mx = fmaxf(mx, acc);
            }
            float cur = mx;
            if (t > 0) {
                float rec = r1b[c];
                for (int ci = 0; ci < HCn; ++ci) rec += s1p[ci * 289 + rem] * r1w[c * HCn + ci];
                cur += rec;
            }
            float mm = m1[idx];
            float reset = mm > 1.0f ? 1.0f : 0.0f;
            mm = 0.9f * mm + cur - reset;
            m1[idx] = mm;
            s1c[idx] = (mm - 1.0f) > 0.f ? 1.f : 0.f;
        }
        __syncthreads();

        // ---- stage 2: conv2(3x3 SAME on 17x17) -> maxpool2 -> +r2(s2_prev) ----
        for (int idx = tid; idx < 2048; idx += NT_) {
            int c = idx >> 6, rem = idx & 63, i = rem >> 3, j = rem & 7;
            float mx = -1e30f;
            for (int di = 0; di < 2; ++di)
            for (int dj = 0; dj < 2; ++dj) {
                int oy = 2 * i + di, ox = 2 * j + dj;
                float acc = c2b[c];
                for (int ci = 0; ci < HCn; ++ci) {
                    const float* sp = s1c + ci * 289;
                    const float* wp = c2w + (c * HCn + ci) * 9;
                    for (int ky = 0; ky < 3; ++ky) {
                        int yy = oy + ky - 1; if ((unsigned)yy >= 17u) continue;
                        for (int kx = 0; kx < 3; ++kx) {
                            int xx = ox + kx - 1; if ((unsigned)xx >= 17u) continue;
                            acc += sp[yy * 17 + xx] * wp[ky * 3 + kx];
                        }
                    }
                }
                mx = fmaxf(mx, acc);
            }
            float rec = 0.f;
            if (t > 0) {
                rec = r2b[c];
                for (int ci = 0; ci < HCn; ++ci) rec += s2[ci * 64 + rem] * r2w[c * HCn + ci];
            }
            tmp[idx] = mx + rec;
        }
        __syncthreads();
        for (int idx = tid; idx < 2048; idx += NT_) {       // LIF 2
            float mm = m2[idx];
            float reset = mm > 1.0f ? 1.0f : 0.0f;
            mm = 0.9f * mm + tmp[idx] - reset;
            m2[idx] = mm;
            s2[idx] = (mm - 1.0f) > 0.f ? 1.f : 0.f;
        }
        __syncthreads();

        // ---- stage 3: conv3(3x3 SAME on 8x8) + r3(s3_prev) ----
        for (int idx = tid; idx < 4096; idx += NT_) {
            int c = idx >> 6, rem = idx & 63, i = rem >> 3, j = rem & 7;
            float acc = c3b[c];
            for (int ci = 0; ci < HCn; ++ci) {
                const float* sp = s2 + ci * 64;
                const float* wp = c3w + (c * HCn + ci) * 9;
                for (int ky = 0; ky < 3; ++ky) {
                    int yy = i + ky - 1; if ((unsigned)yy >= 8u) continue;
                    for (int kx = 0; kx < 3; ++kx) {
                        int xx = j + kx - 1; if ((unsigned)xx >= 8u) continue;
                        acc += sp[yy * 8 + xx] * wp[ky * 3 + kx];
                    }
                }
            }
            float rec = 0.f;
            if (t > 0) {
                rec = r3b[c];
                for (int ci = 0; ci < SCn; ++ci) rec += s3[ci * 64 + rem] * r3w[c * SCn + ci];
            }
            tmp[idx] = acc + rec;
        }
        __syncthreads();
        for (int idx = tid; idx < 4096; idx += NT_) {       // LIF 3 + spike out
            float mm = m3[idx];
            float reset = mm > 1.0f ? 1.0f : 0.0f;
            mm = 0.9f * mm + tmp[idx] - reset;
            m3[idx] = mm;
            float spk = (mm - 1.0f) > 0.f ? 1.f : 0.f;
            s3[idx] = spk;
            spikes[(size_t)(b * Tn + t) * SFEAT + idx] = spk;
        }
        __syncthreads();
    }
}

// ============================================================
// spike_to_activation: 5-step sliding mean over T, emit f16
// ============================================================
__global__ void acts_kernel(const float* __restrict__ spikes, _Float16* __restrict__ acts16) {
    int idx = blockIdx.x * blockDim.x + threadIdx.x;   // over B*SFEAT
    if (idx >= Bn * SFEAT) return;
    int b = idx >> 12, f = idx & (SFEAT - 1);
    float sum = 0.f;
    for (int t = 0; t < Tn; ++t) {
        sum += spikes[(size_t)(b * Tn + t) * SFEAT + f];
        if (t >= 5) sum -= spikes[(size_t)(b * Tn + t - 5) * SFEAT + f];
        acts16[(size_t)(b * Tn + t) * SFEAT + f] = (_Float16)(sum * 0.2f);
    }
}

// ============================================================
// WMMA GEMM: out(M,N) = A(M,K) * W(N,K)^T  [+ bias] [+= out]
// 8 waves/block; each wave computes a 16 x (16*NT) tile: the A fragment
// is loaded once per K-step and fed to NT WMMAs (arithmetic-intensity
// blocking on the bandwidth-bound operand). K step 32 (f16 WMMA).
// Lane layout per CDNA5 ISA 7.12.2 (16-bit A 16x32, B 32x16).
// ============================================================
template <int NT>
__global__ void gemm_wmma_f16(const _Float16* __restrict__ A,
                              const _Float16* __restrict__ W,
                              const float* __restrict__ bias,
                              float* __restrict__ out,
                              int M, int N, int K, int add_out)
{
    const int lane = threadIdx.x & 31;
    const int wave = threadIdx.x >> 5;
    const int m0 = blockIdx.x * 128 + wave * 16;
    const int n0 = blockIdx.y * (16 * NT);
    if (m0 >= M) return;

    const int r  = lane & 15;
    const int hi = lane >> 4;
    const _Float16* arow = A + (size_t)(m0 + r) * K;   // A row (M index)
    const _Float16* brow[NT];
#pragma unroll
    for (int tn = 0; tn < NT; ++tn)
        brow[tn] = W + (size_t)(n0 + tn * 16 + r) * K; // B column = W row (N index)

    v8f acc[NT];
#pragma unroll
    for (int tn = 0; tn < NT; ++tn)
        acc[tn] = (v8f){0.f, 0.f, 0.f, 0.f, 0.f, 0.f, 0.f, 0.f};

    for (int k0 = 0; k0 < K; k0 += 32) {
        __builtin_prefetch(arow + k0 + 256, 0, 1);   // global_prefetch path
        // A 16x32: low lanes hold K {0..7, 16..23}; high lanes {8..15, 24..31}
        v8h a0 = *(const v8h*)(arow + k0 + hi * 8);
        v8h a1 = *(const v8h*)(arow + k0 + 16 + hi * 8);
        v16h av = __builtin_shufflevector(a0, a1, 0,1,2,3,4,5,6,7,8,9,10,11,12,13,14,15);
#pragma unroll
        for (int tn = 0; tn < NT; ++tn) {
            // B 32x16: low lanes hold K {0..7, 16..23}; high lanes {8..15, 24..31} of their column
            v8h b0 = *(const v8h*)(brow[tn] + k0 + hi * 8);
            v8h b1 = *(const v8h*)(brow[tn] + k0 + 16 + hi * 8);
            v16h bv = __builtin_shufflevector(b0, b1, 0,1,2,3,4,5,6,7,8,9,10,11,12,13,14,15);
            acc[tn] = __builtin_amdgcn_wmma_f32_16x16x32_f16(false, av, false, bv,
                                                             (short)0, acc[tn], false, false);
        }
    }

#pragma unroll
    for (int tn = 0; tn < NT; ++tn) {
        const int col = n0 + tn * 16 + (lane & 15);
        const float bv2 = bias ? bias[col] : 0.0f;
#pragma unroll
        for (int rr = 0; rr < 8; ++rr) {
            int row = m0 + rr + hi * 8;             // C/D: VGPR rr -> M = rr + 8*hi
            size_t o = (size_t)row * N + col;
            float v = acc[tn][rr] + bv2;
            if (add_out) out[o] += v; else out[o] = v;
        }
    }
}

// ============================================================
// LayerNorm over D=128, output f16 (one block per row)
// ============================================================
__global__ void ln_f16_kernel(const float* __restrict__ h,
                              const float* __restrict__ g, const float* __restrict__ bb,
                              _Float16* __restrict__ out)
{
    __shared__ float red[DM];
    const int m = blockIdx.x, d = threadIdx.x;
    float v = h[(size_t)m * DM + d];
    red[d] = v; __syncthreads();
    for (int s = 64; s > 0; s >>= 1) { if (d < s) red[d] += red[d + s]; __syncthreads(); }
    float mean = red[0] * (1.0f / DM); __syncthreads();
    float c = v - mean;
    red[d] = c * c; __syncthreads();
    for (int s = 64; s > 0; s >>= 1) { if (d < s) red[d] += red[d + s]; __syncthreads(); }
    float var = red[0] * (1.0f / DM);
    out[(size_t)m * DM + d] = (_Float16)(c * rsqrtf(var + 1e-5f) * g[d] + bb[d]);
}

// ============================================================
// Depthwise causal conv1d (k=4) + SiLU on first half of xz
// ============================================================
__global__ void dwconv_silu_kernel(const float* __restrict__ xz,
                                   const float* __restrict__ convw,
                                   float* __restrict__ xc32, _Float16* __restrict__ xc16)
{
    int idx = blockIdx.x * blockDim.x + threadIdx.x;   // over MTOT*DI
    if (idx >= MTOT * DI) return;
    int m = idx >> 8, d = idx & (DI - 1);
    int b = m / Tn, l = m % Tn;
    float s = 0.f;
#pragma unroll
    for (int kk = 0; kk < 4; ++kk) {
        int lp = l + kk - 3;
        if (lp >= 0) s += xz[(size_t)(b * Tn + lp) * (2 * DI) + d] * convw[d * 4 + kk];
    }
    s = s / (1.0f + __expf(-s));   // SiLU
    xc32[(size_t)m * DI + d] = s;
    xc16[(size_t)m * DI + d] = (_Float16)s;
}

// ============================================================
// x_proj: xdbl(M,40) = xc(M,256) @ xproj_w(40,256)^T   (small, VALU)
// ============================================================
__global__ void xproj_kernel(const float* __restrict__ xc,
                             const float* __restrict__ xprojw,
                             float* __restrict__ xdbl)
{
    int idx = blockIdx.x * blockDim.x + threadIdx.x;   // over MTOT*40
    if (idx >= MTOT * (DTR + 2 * DS)) return;
    int m = idx / 40, j = idx % 40;
    const float* xr = xc + (size_t)m * DI;
    const float* wr = xprojw + j * DI;
    float s = 0.f;
    for (int k = 0; k < DI; ++k) s += xr[k] * wr[k];
    xdbl[(size_t)m * 40 + j] = s;
}

// ============================================================
// dt = softplus(xdbl[:, :8] @ dt_w^T + dt_b)
// ============================================================
__global__ void dt_kernel(const float* __restrict__ xdbl,
                          const float* __restrict__ dtw, const float* __restrict__ dtb,
                          float* __restrict__ dt)
{
    int idx = blockIdx.x * blockDim.x + threadIdx.x;   // over MTOT*DI
    if (idx >= MTOT * DI) return;
    int m = idx >> 8, d = idx & (DI - 1);
    float s = dtb[d];
#pragma unroll
    for (int k = 0; k < DTR; ++k) s += xdbl[(size_t)m * 40 + k] * dtw[d * DTR + k];
    s = (s > 20.0f) ? s : log1pf(__expf(s));
    dt[(size_t)m * DI + d] = s;
}

// ============================================================
// Selective scan + skip + gate; output y as f16 for out_proj
// One thread per (b, d); 16 SSM states in registers, loop L=128.
// ============================================================
__global__ void scan_kernel(const float* __restrict__ dt,
                            const float* __restrict__ xc32,
                            const float* __restrict__ xdbl,
                            const float* __restrict__ xz,
                            const float* __restrict__ Alog,
                            const float* __restrict__ Dp,
                            _Float16* __restrict__ y16)
{
    int idx = blockIdx.x * blockDim.x + threadIdx.x;   // over B*DI
    if (idx >= Bn * DI) return;
    int b = idx >> 8, d = idx & (DI - 1);

    float Ar[DS], st[DS];
#pragma unroll
    for (int n = 0; n < DS; ++n) { Ar[n] = -__expf(Alog[d * DS + n]); st[n] = 0.f; }
    const float Dpv = Dp[d];

    for (int l = 0; l < Tn; ++l) {
        int m = b * Tn + l;
        float dtv = dt[(size_t)m * DI + d];
        float u   = xc32[(size_t)m * DI + d];
        const float* row = xdbl + (size_t)m * 40;
        float y = 0.f;
#pragma unroll
        for (int n = 0; n < DS; ++n) {
            float Bv = row[8 + n];
            float Cv = row[24 + n];
            st[n] = __expf(dtv * Ar[n]) * st[n] + dtv * Bv * u;
            y += st[n] * Cv;
        }
        y += u * Dpv;
        float res = xz[(size_t)m * (2 * DI) + DI + d];
        float gate = res / (1.0f + __expf(-res));
        y16[(size_t)m * DI + d] = (_Float16)(y * gate);
    }
}

// ============================================================
// Final: mean over T, layernorm, classifier (one block per b)
// ============================================================
__global__ void final_kernel(const float* __restrict__ h,
                             const float* __restrict__ fng, const float* __restrict__ fnb,
                             const float* __restrict__ clsw, const float* __restrict__ clsb,
                             float* __restrict__ out)
{
    __shared__ float red[DM];
    __shared__ float hn[DM];
    const int b = blockIdx.x, d = threadIdx.x;
    float s = 0.f;
    for (int t = 0; t < Tn; ++t) s += h[(size_t)(b * Tn + t) * DM + d];
    s *= (1.0f / Tn);
    red[d] = s; __syncthreads();
    for (int k = 64; k > 0; k >>= 1) { if (d < k) red[d] += red[d + k]; __syncthreads(); }
    float mean = red[0] * (1.0f / DM); __syncthreads();
    float c = s - mean;
    red[d] = c * c; __syncthreads();
    for (int k = 64; k > 0; k >>= 1) { if (d < k) red[d] += red[d + k]; __syncthreads(); }
    float var = red[0] * (1.0f / DM);
    hn[d] = c * rsqrtf(var + 1e-5f) * fng[d] + fnb[d];
    __syncthreads();
    if (d < 10) {
        float o = clsb[d];
        for (int k = 0; k < DM; ++k) o += hn[k] * clsw[d * DM + k];
        out[b * 10 + d] = o;
    }
}

// ============================================================
// Host launcher
// ============================================================
extern "C" void kernel_launch(void* const* d_in, const int* in_sizes, int n_in,
                              void* d_out, int out_size, void* d_ws, size_t ws_size,
                              hipStream_t stream) {
    (void)in_sizes; (void)n_in; (void)out_size; (void)ws_size;

    const float* x      = (const float*)d_in[0];
    const float* c1w    = (const float*)d_in[1];
    const float* c1b    = (const float*)d_in[2];
    const float* c2w    = (const float*)d_in[3];
    const float* c2b    = (const float*)d_in[4];
    const float* c3w    = (const float*)d_in[5];
    const float* c3b    = (const float*)d_in[6];
    const float* r1w    = (const float*)d_in[7];
    const float* r1b    = (const float*)d_in[8];
    const float* r2w    = (const float*)d_in[9];
    const float* r2b    = (const float*)d_in[10];
    const float* r3w    = (const float*)d_in[11];
    const float* r3b    = (const float*)d_in[12];
    const float* proj_w = (const float*)d_in[13];
    const float* proj_b = (const float*)d_in[14];
    const float* ln_g   = (const float*)d_in[15];
    const float* ln_b   = (const float*)d_in[16];
    const float* in_w   = (const float*)d_in[17];
    const float* conv_w = (const float*)d_in[18];
    const float* xproj_w= (const float*)d_in[19];
    const float* dt_w   = (const float*)d_in[20];
    const float* dt_b   = (const float*)d_in[21];
    const float* Alog   = (const float*)d_in[22];
    const float* Dp     = (const float*)d_in[23];
    const float* out_w  = (const float*)d_in[24];
    const float* fn_g   = (const float*)d_in[25];
    const float* fn_b   = (const float*)d_in[26];
    const float* cls_w  = (const float*)d_in[27];
    const float* cls_b  = (const float*)d_in[28];
    float* out = (float*)d_out;

    // ---- carve workspace (256B aligned) ----
    char* base = (char*)d_ws;
    size_t off = 0;
    auto carve = [&](size_t bytes) -> char* {
        char* p = base + off;
        off = (off + bytes + 255) & ~(size_t)255;
        return p;
    };
    float*    spikes = (float*)   carve((size_t)Bn * Tn * SFEAT * 4);
    _Float16* acts16 = (_Float16*)carve((size_t)MTOT * SFEAT * 2);
    _Float16* pw16   = (_Float16*)carve((size_t)DM * SFEAT * 2);
    _Float16* inw16  = (_Float16*)carve((size_t)4 * 2 * DI * DM * 2);
    _Float16* outw16 = (_Float16*)carve((size_t)4 * DM * DI * 2);
    float*    h      = (float*)   carve((size_t)MTOT * DM * 4);
    _Float16* xn16   = (_Float16*)carve((size_t)MTOT * DM * 2);
    float*    xz     = (float*)   carve((size_t)MTOT * 2 * DI * 4);
    float*    xc32   = (float*)   carve((size_t)MTOT * DI * 4);
    _Float16* xc16   = (_Float16*)carve((size_t)MTOT * DI * 2);
    float*    xdbl   = (float*)   carve((size_t)MTOT * 40 * 4);
    float*    dtb_   = (float*)   carve((size_t)MTOT * DI * 4);
    _Float16* y16    = (_Float16*)carve((size_t)MTOT * DI * 2);
    float*    s1ga   = (float*)   carve((size_t)Bn * 9248 * 4);
    float*    s1gb   = (float*)   carve((size_t)Bn * 9248 * 4);
    float*    mem1   = (float*)   carve((size_t)Bn * 9248 * 4);
    float*    mem2   = (float*)   carve((size_t)Bn * 2048 * 4);
    float*    mem3   = (float*)   carve((size_t)Bn * 4096 * 4);

    // ---- stage weights to f16 ----
    {
        int n = DM * SFEAT;
        cvt_f32_f16<<<(n + 255) / 256, 256, 0, stream>>>(proj_w, pw16, n);
        n = 4 * 2 * DI * DM;
        cvt_f32_f16<<<(n + 255) / 256, 256, 0, stream>>>(in_w, inw16, n);
        n = 4 * DM * DI;
        cvt_f32_f16<<<(n + 255) / 256, 256, 0, stream>>>(out_w, outw16, n);
    }

    // ---- spiking frontend: B workgroups, T-loop inside ----
    frontend_kernel<<<Bn, 512, 0, stream>>>(x, c1w, c1b, c2w, c2b, c3w, c3b,
                                            r1w, r1b, r2w, r2b, r3w, r3b,
                                            s1ga, s1gb, mem1, mem2, mem3, spikes);

    // ---- sliding-window activation (f16) ----
    acts_kernel<<<(Bn * SFEAT) / 256, 256, 0, stream>>>(spikes, acts16);

    // ---- projection GEMM (WMMA): h = acts @ proj_w^T + proj_b ----
    {
        dim3 grid(MTOT / 128, DM / 32);          // 64 x 4 blocks, 16x32 per wave
        gemm_wmma_f16<2><<<grid, 256, 0, stream>>>(acts16, pw16, proj_b, h,
                                                   MTOT, DM, SFEAT, 0);
    }

    // ---- 4 Mamba layers ----
    for (int i = 0; i < 4; ++i) {
        ln_f16_kernel<<<MTOT, DM, 0, stream>>>(h, ln_g + i * DM, ln_b + i * DM, xn16);

        {   // in_proj: xz(M,512) = xn @ in_w^T   (16x64 per wave)
            dim3 grid(MTOT / 128, (2 * DI) / 64);
            gemm_wmma_f16<4><<<grid, 256, 0, stream>>>(xn16, inw16 + (size_t)i * 2 * DI * DM,
                                                       nullptr, xz, MTOT, 2 * DI, DM, 0);
        }

        dwconv_silu_kernel<<<(MTOT * DI) / 256, 256, 0, stream>>>(
            xz, conv_w + (size_t)i * DI * 4, xc32, xc16);

        xproj_kernel<<<(MTOT * 40 + 255) / 256, 256, 0, stream>>>(
            xc32, xproj_w + (size_t)i * 40 * DI, xdbl);

        dt_kernel<<<(MTOT * DI) / 256, 256, 0, stream>>>(
            xdbl, dt_w + (size_t)i * DI * DTR, dt_b + (size_t)i * DI, dtb_);

        scan_kernel<<<(Bn * DI) / 256, 256, 0, stream>>>(
            dtb_, xc32, xdbl, xz, Alog + (size_t)i * DI * DS, Dp + (size_t)i * DI, y16);

        {   // out_proj (WMMA), accumulate residual into h  (16x32 per wave)
            dim3 grid(MTOT / 128, DM / 32);
            gemm_wmma_f16<2><<<grid, 256, 0, stream>>>(y16, outw16 + (size_t)i * DM * DI,
                                                       nullptr, h, MTOT, DM, DI, 1);
        }
    }

    // ---- final mean + LN + classifier ----
    final_kernel<<<Bn, DM, 0, stream>>>(h, fn_g, fn_b, cls_w, cls_b, out);
}
